// SkimRNN_66090956751406
// MI455X (gfx1250) — compile-verified, matrix operand order checked by reading, not run on hardware
//
#include <hip/hip_runtime.h>
#include <hip/hip_bf16.h>
#include <math.h>

// ---------------------------------------------------------------------------
// SkimRNN on MI455X (gfx1250): bf16 WMMA recurrent GEMMs, fused gate epilogue,
// async global->LDS A-staging, register-double-buffered B fragments with
// sched_barrier-enforced software pipelining.
// B=64, L=512, E=512, HL=1024, HS=256, V=32000, NC=5.
// ---------------------------------------------------------------------------

typedef __attribute__((ext_vector_type(16))) __bf16 v16bf;
typedef __attribute__((ext_vector_type(8)))  float  v8f;

#define Bv   64
#define Lv   512
#define Ev   512
#define HLv  1024
#define HSv  256
#define NCv  5
#define KLv  (Ev + HLv)   // 1536  (concat(e_t, h_large))
#define KSv  (Ev + HSv)   // 768   (concat(e_t, h_small))
#define KCHUNK 256        // A-matrix K staging chunk in LDS (64*256*2B = 32 KB)

__device__ __forceinline__ float sigm(float x) { return 1.0f / (1.0f + __expf(-x)); }

// Wait for all outstanding async (global<->LDS) transfers of this wave.
__device__ __forceinline__ void wait_async0() {
#if __has_builtin(__builtin_amdgcn_s_wait_asynccnt)
  __builtin_amdgcn_s_wait_asynccnt(0);
#else
  asm volatile("s_wait_asynccnt 0x0" ::: "memory");
#endif
}

// Async copy 16 bytes global -> LDS (bypasses VGPRs; tracked by ASYNCcnt).
__device__ __forceinline__ void async_copy16(const void* gsrc, void* lds_dst) {
  unsigned ldsoff = (unsigned)(size_t)lds_dst;  // LDS aperture: offset in [31:0]
  asm volatile("global_load_async_to_lds_b128 %0, %1, off"
               :: "v"(ldsoff), "v"(gsrc) : "memory");
}

// Hard scheduling fence: nothing may be reordered across this point.
// Used to pin "preload next fragments" above "WMMA on current fragments",
// so the machine scheduler cannot sink loads to their use sites and
// re-serialize the load->WMMA pipeline.
__device__ __forceinline__ void sched_split() {
#if __has_builtin(__builtin_amdgcn_sched_barrier)
  __builtin_amdgcn_sched_barrier(0);
#endif
}

// ---------------------------------------------------------------------------
// Prep kernels (run once per call; all parallel, no sequential dependence)
// ---------------------------------------------------------------------------

// Gather embedding rows by token id, convert to bf16, time-major [L][B][E].
__global__ __launch_bounds__(256) void prep_embed(const int* __restrict__ x,
                                                  const float* __restrict__ emb,
                                                  __bf16* __restrict__ Xbf) {
  size_t i = (size_t)blockIdx.x * blockDim.x + threadIdx.x;
  size_t total = (size_t)Lv * Bv * Ev;
  if (i >= total) return;
  int e = (int)(i % Ev);
  int b = (int)((i / Ev) % Bv);
  int t = (int)(i / ((size_t)Ev * Bv));
  int tok = x[b * Lv + t];
  Xbf[i] = (__bf16)emb[(size_t)tok * Ev + e];
}

// Concatenate W_ih | W_hh along K into one bf16 matrix [4H][Ev + H].
__global__ __launch_bounds__(256) void prep_w(const float* __restrict__ Wih,
                                              const float* __restrict__ Whh,
                                              int rows, int K,
                                              __bf16* __restrict__ Wout) {
  size_t i = (size_t)blockIdx.x * blockDim.x + threadIdx.x;
  if (i >= (size_t)rows * K) return;
  int k = (int)(i % K);
  int n = (int)(i / K);
  float v = (k < Ev) ? Wih[(size_t)n * Ev + k]
                     : Whh[(size_t)n * (K - Ev) + (k - Ev)];
  Wout[i] = (__bf16)v;
}

__global__ __launch_bounds__(256) void prep_bias(const float* __restrict__ bih,
                                                 const float* __restrict__ bhh,
                                                 int n, float* __restrict__ out) {
  int i = blockIdx.x * blockDim.x + threadIdx.x;
  if (i < n) out[i] = bih[i] + bhh[i];
}

// ---------------------------------------------------------------------------
// Per-step LSTM gates + cell update (both cells in one launch).
// Blocks 0..31: large cell (32 hidden units each).  Blocks 32..39: small cell.
// Each workgroup owns hidden units [jbase, jbase+32) and computes the i/f/g/o
// column groups for them, so the LSTM nonlinearity fuses into the GEMM
// epilogue with no extra global pass.
// Wave w (of 8): M-tile = w&3 (16 batch rows), j-subtile = w>>2 (16 units),
// 4 f32 16x16 accumulators (one per gate group), A fragment reused 4x.
// A is staged global->LDS with async-to-LDS; B fragments are register
// double-buffered across K iterations; sched_barrier keeps the preloads
// ahead of the WMMA block so VMEM latency overlaps the matrix pipes.
// ---------------------------------------------------------------------------
__global__ __launch_bounds__(256) void lstm_gates_step(
    int t, const __bf16* __restrict__ Xbf,
    const __bf16* __restrict__ Wl, const float* __restrict__ bl,
    const __bf16* __restrict__ hbfl, const float* __restrict__ c_l,
    float* __restrict__ hLn, float* __restrict__ cLn,
    const __bf16* __restrict__ Ws, const float* __restrict__ bs,
    const __bf16* __restrict__ hbfs, const float* __restrict__ c_s,
    float* __restrict__ hSn, float* __restrict__ cSn) {
  __shared__ __align__(32) __bf16 As[Bv * KCHUNK];  // 32 KB A-chunk stage

  const bool big = blockIdx.x < (HLv / 32);
  const int  jbase = big ? (int)blockIdx.x * 32 : ((int)blockIdx.x - HLv / 32) * 32;
  const int  H = big ? HLv : HSv;
  const int  K = big ? KLv : KSv;
  const __bf16* W    = big ? Wl   : Ws;
  const float*  bias = big ? bl   : bs;
  const __bf16* hbf  = big ? hbfl : hbfs;
  const float*  cp   = big ? c_l  : c_s;
  float* hn = big ? hLn : hSn;
  float* cn = big ? cLn : cSn;

  const int tid   = threadIdx.x;
  const int lane  = tid & 31;
  const int wave  = tid >> 5;
  const int mtile = wave & 3;    // 4 M-tiles cover B=64
  const int half  = wave >> 2;   // 2 j-subtiles cover the 32-unit slice
  const int col   = lane & 15;
  const int khalf = lane >> 4;
  const int jcol  = jbase + half * 16 + col;  // this lane's hidden-unit column

  // Per-gate weight-row base pointers (row-major [4H][K]).
  const __bf16* wrow[4];
#pragma unroll
  for (int g = 0; g < 4; ++g)
    wrow[g] = W + (size_t)(g * H + jcol) * K + khalf * 16;
  const __bf16* arow = As + (mtile * 16 + col) * KCHUNK + khalf * 16;

  v8f acc[4] = {};  // i, f, g, o accumulators (f32)

  for (int kb0 = 0; kb0 < K; kb0 += KCHUNK) {
    // Stage A = concat(e_t, h_prev)[64][kb0:kb0+KCHUNK] via async global->LDS.
    const int cpr = KCHUNK / 8;  // 8-bf16 (16B) chunks per row
    for (int i = tid; i < Bv * cpr; i += 256) {
      int r  = i / cpr;
      int cc = kb0 + (i % cpr) * 8;
      const __bf16* src = (cc < Ev)
          ? (Xbf + ((size_t)t * Bv + r) * Ev + cc)
          : (hbf + r * H + (cc - Ev));
      async_copy16(src, As + r * KCHUNK + (cc - kb0));
    }
    wait_async0();     // this wave's async transfers done
    __syncthreads();   // everyone's transfers visible in LDS

    // Prefetch next weight chunk for this lane's column (keeps L2 hot).
    __builtin_prefetch(W + (size_t)jcol * K + kb0 + KCHUNK, 0, 1);

    // Software pipeline: preload fragments for kk+32, then WMMA on kk.
    v16bf a_cur = *(const v16bf*)(arow + 0);
    v16bf b_cur[4];
#pragma unroll
    for (int g = 0; g < 4; ++g) b_cur[g] = *(const v16bf*)(wrow[g] + kb0);

#pragma unroll
    for (int kk = 0; kk < KCHUNK; kk += 32) {
      v16bf a_nxt = a_cur;
      v16bf b_nxt[4] = {b_cur[0], b_cur[1], b_cur[2], b_cur[3]};
      const int kn = kk + 32;
      if (kn < KCHUNK) {  // compile-time under full unroll
        a_nxt = *(const v16bf*)(arow + kn);
#pragma unroll
        for (int g = 0; g < 4; ++g)
          b_nxt[g] = *(const v16bf*)(wrow[g] + kb0 + kn);
      }
      // Pin the preloads above the WMMA block: the scheduler may not sink
      // next-iteration loads past this point, so their latency overlaps the
      // 4 WMMAs below instead of serializing in front of them.
      sched_split();
#pragma unroll
      for (int g = 0; g < 4; ++g)
        acc[g] = __builtin_amdgcn_wmma_f32_16x16x32_bf16(
            false, a_cur, false, b_cur[g], (short)0, acc[g], false, false);
      a_cur = a_nxt;
#pragma unroll
      for (int g = 0; g < 4; ++g) b_cur[g] = b_nxt[g];
    }
    __syncthreads();  // before re-staging As
  }

  // Epilogue: C/D layout => VGPR e holds M = mtile*16 + khalf*8 + e, N = jcol.
  const int rbase = mtile * 16 + khalf * 8;
  const float bi = bias[0 * H + jcol], bf_ = bias[1 * H + jcol];
  const float bg = bias[2 * H + jcol], bo  = bias[3 * H + jcol];
#pragma unroll
  for (int e = 0; e < 8; ++e) {
    int m = rbase + e;
    float gi = acc[0][e] + bi;
    float gf = acc[1][e] + bf_;
    float gg = acc[2][e] + bg;
    float go = acc[3][e] + bo;
    float c2 = sigm(gf) * cp[m * H + jcol] + sigm(gi) * tanhf(gg);
    float h2 = sigm(go) * tanhf(c2);
    hn[m * H + jcol] = h2;
    cn[m * H + jcol] = c2;
  }
}

// ---------------------------------------------------------------------------
// Per-step blend + skim projection.
// Blocks 0..255: elementwise gumbel blend over [64][1024]; updates states,
//                writes h_stack[:, t, :].  Blocks 256..319: p_t per batch row.
// ---------------------------------------------------------------------------
__global__ __launch_bounds__(256) void blend_step(
    int t, const float* __restrict__ gum, const __bf16* __restrict__ Xbf,
    const float* __restrict__ hLn, const float* __restrict__ cLn,
    const float* __restrict__ hSn, const float* __restrict__ cSn,
    float* __restrict__ h_l, float* __restrict__ c_l, __bf16* __restrict__ hbfl,
    float* __restrict__ h_s, float* __restrict__ c_s, __bf16* __restrict__ hbfs,
    const float* __restrict__ W_p, const float* __restrict__ b_p,
    float* __restrict__ out_h, float* __restrict__ out_p) {
  const int blk = blockIdx.x, tid = threadIdx.x;
  if (blk < 256) {
    int idx = blk * 256 + tid;
    int b = idx >> 10, j = idx & 1023;
    float u0 = gum[((size_t)b * Lv + t) * 2 + 0];
    float u1 = gum[((size_t)b * Lv + t) * 2 + 1];
    float g0 = -__logf(-__logf(u0));
    float g1 = -__logf(-__logf(u1));
    float mx = fmaxf(g0, g1);
    float e0 = __expf(g0 - mx), e1 = __expf(g1 - mx);
    float inv = 1.0f / (e0 + e1);
    float r0 = e0 * inv, r1 = e1 * inv;
    float hlN = hLn[idx], clN = cLn[idx];
    float hb = (j < HSv) ? hSn[b * HSv + j] : h_l[idx];   // old carry tail
    float cb = (j < HSv) ? cSn[b * HSv + j] : clN;
    float hnext = r0 * hlN + r1 * hb;
    float cnext = r0 * clN + r1 * cb;
    h_l[idx] = hnext;
    c_l[idx] = cnext;
    hbfl[idx] = (__bf16)hnext;
    out_h[((size_t)b * Lv + t) * HLv + j] = hnext;
    if (j < HSv) {  // commit small-cell state for next step
      float hs = hSn[b * HSv + j];
      h_s[b * HSv + j] = hs;
      c_s[b * HSv + j] = cSn[b * HSv + j];
      hbfs[b * HSv + j] = (__bf16)hs;
    }
  } else {
    // p_t[b] = softmax( concat(e_t, h_l_new, c_l_new) @ W_p.T + b_p )
    int b = blk - 256;
    __shared__ float s0[256], s1[256];
    const int P = Ev + 2 * HLv;  // 2560
    float d0 = 0.f, d1 = 0.f;
    for (int k = tid; k < P; k += 256) {
      float v;
      if (k < Ev)            v = (float)Xbf[((size_t)t * Bv + b) * Ev + k];
      else if (k < Ev + HLv) v = hLn[b * HLv + (k - Ev)];
      else                   v = cLn[b * HLv + (k - Ev - HLv)];
      d0 += v * W_p[k];
      d1 += v * W_p[P + k];
    }
    s0[tid] = d0; s1[tid] = d1;
    __syncthreads();
    for (int s = 128; s > 0; s >>= 1) {
      if (tid < s) { s0[tid] += s0[tid + s]; s1[tid] += s1[tid + s]; }
      __syncthreads();
    }
    if (tid == 0) {
      float p0 = s0[0] + b_p[0], p1 = s1[0] + b_p[1];
      float mx = fmaxf(p0, p1);
      float e0 = __expf(p0 - mx), e1 = __expf(p1 - mx);
      float inv = 1.0f / (e0 + e1);
      out_p[((size_t)b * Lv + t) * 2 + 0] = e0 * inv;
      out_p[((size_t)b * Lv + t) * 2 + 1] = e1 * inv;
    }
  }
}

// ---------------------------------------------------------------------------
// Final classifier: logits = softmax(relu(relu(h_l)@W1.T+b1)@W2.T+b2). Tiny.
// ---------------------------------------------------------------------------
__global__ __launch_bounds__(256) void classifier(
    const float* __restrict__ h_l, const float* __restrict__ W1,
    const float* __restrict__ b1, const float* __restrict__ W2,
    const float* __restrict__ b2, float* __restrict__ out_logits) {
  const int b = blockIdx.x, tid = threadIdx.x;
  __shared__ float z[512];
  __shared__ float lg[NCv];
  for (int jj = tid; jj < 512; jj += 256) {
    float acc = b1[jj];
    for (int k = 0; k < HLv; ++k)
      acc += fmaxf(h_l[b * HLv + k], 0.f) * W1[(size_t)jj * HLv + k];
    z[jj] = fmaxf(acc, 0.f);
  }
  __syncthreads();
  if (tid < NCv) {
    float acc = b2[tid];
    for (int k = 0; k < 512; ++k) acc += z[k] * W2[tid * 512 + k];
    lg[tid] = acc;
  }
  __syncthreads();
  if (tid == 0) {
    float mx = lg[0];
    for (int o = 1; o < NCv; ++o) mx = fmaxf(mx, lg[o]);
    float s = 0.f, e[NCv];
    for (int o = 0; o < NCv; ++o) { e[o] = __expf(lg[o] - mx); s += e[o]; }
    for (int o = 0; o < NCv; ++o) out_logits[b * NCv + o] = e[o] / s;
  }
}

// ---------------------------------------------------------------------------
extern "C" void kernel_launch(void* const* d_in, const int* in_sizes, int n_in,
                              void* d_out, int out_size, void* d_ws, size_t ws_size,
                              hipStream_t stream) {
  const int*   x     = (const int*)d_in[0];
  const float* gum   = (const float*)d_in[1];
  const float* emb   = (const float*)d_in[2];
  const float* Wih_l = (const float*)d_in[3];
  const float* Whh_l = (const float*)d_in[4];
  const float* bih_l = (const float*)d_in[5];
  const float* bhh_l = (const float*)d_in[6];
  const float* Wih_s = (const float*)d_in[7];
  const float* Whh_s = (const float*)d_in[8];
  const float* bih_s = (const float*)d_in[9];
  const float* bhh_s = (const float*)d_in[10];
  const float* W_p   = (const float*)d_in[11];
  const float* b_p   = (const float*)d_in[12];
  const float* W_c1  = (const float*)d_in[13];
  const float* b_c1  = (const float*)d_in[14];
  const float* W_c2  = (const float*)d_in[15];
  const float* b_c2  = (const float*)d_in[16];

  float* out_logits = (float*)d_out;
  float* out_h = out_logits + Bv * NCv;
  float* out_p = out_h + (size_t)Bv * Lv * HLv;

  // Workspace carve-out (all 256B aligned), ~50 MB total.
  char* ws = (char*)d_ws;
  size_t off = 0;
  auto carve = [&](size_t bytes) -> void* {
    void* p = ws + off;
    off += (bytes + 255) & ~(size_t)255;
    return p;
  };
  __bf16* Xbf = (__bf16*)carve((size_t)Lv * Bv * Ev * 2);     // 32 MB
  __bf16* Wl  = (__bf16*)carve((size_t)4 * HLv * KLv * 2);    // 12 MB
  __bf16* Wsm = (__bf16*)carve((size_t)4 * HSv * KSv * 2);    // 1.5 MB
  float*  bl  = (float*)carve(4 * HLv * 4);
  float*  bs  = (float*)carve(4 * HSv * 4);
  char* state0 = ws + off;
  float*  h_l  = (float*)carve(Bv * HLv * 4);
  float*  c_l  = (float*)carve(Bv * HLv * 4);
  __bf16* hbfl = (__bf16*)carve(Bv * HLv * 2);
  float*  h_s  = (float*)carve(Bv * HSv * 4);
  float*  c_s  = (float*)carve(Bv * HSv * 4);
  __bf16* hbfs = (__bf16*)carve(Bv * HSv * 2);
  size_t stateBytes = (size_t)((ws + off) - state0);
  float* hLn = (float*)carve(Bv * HLv * 4);
  float* cLn = (float*)carve(Bv * HLv * 4);
  float* hSn = (float*)carve(Bv * HSv * 4);
  float* cSn = (float*)carve(Bv * HSv * 4);

  // Zero initial LSTM states (deterministic per call; capture-legal).
  hipMemsetAsync(state0, 0, stateBytes, stream);

  // One-time prep: embedding gather + bf16 weight packs + combined biases.
  {
    size_t total = (size_t)Lv * Bv * Ev;
    prep_embed<<<(unsigned)((total + 255) / 256), 256, 0, stream>>>(x, emb, Xbf);
    size_t twl = (size_t)4 * HLv * KLv;
    prep_w<<<(unsigned)((twl + 255) / 256), 256, 0, stream>>>(Wih_l, Whh_l, 4 * HLv, KLv, Wl);
    size_t tws = (size_t)4 * HSv * KSv;
    prep_w<<<(unsigned)((tws + 255) / 256), 256, 0, stream>>>(Wih_s, Whh_s, 4 * HSv, KSv, Wsm);
    prep_bias<<<(4 * HLv + 255) / 256, 256, 0, stream>>>(bih_l, bhh_l, 4 * HLv, bl);
    prep_bias<<<(4 * HSv + 255) / 256, 256, 0, stream>>>(bih_s, bhh_s, 4 * HSv, bs);
  }

  // Sequential recurrence: 2 launches per step (graph-captured by harness).
  for (int t = 0; t < Lv; ++t) {
    lstm_gates_step<<<HLv / 32 + HSv / 32, 256, 0, stream>>>(
        t, Xbf, Wl, bl, hbfl, c_l, hLn, cLn, Wsm, bs, hbfs, c_s, hSn, cSn);
    blend_step<<<256 + Bv, 256, 0, stream>>>(
        t, gum, Xbf, hLn, cLn, hSn, cSn, h_l, c_l, hbfl, h_s, c_s, hbfs,
        W_p, b_p, out_h, out_p);
  }

  classifier<<<Bv, 256, 0, stream>>>(h_l, W_c1, b_c1, W_c2, b_c2, out_logits);
}
